// CrossModalTranslator_48799418417677
// MI455X (gfx1250) — matrix operand order repaired and checked by reading, not compiled, not run on hardware
//
#include <hip/hip_runtime.h>

// ---------------------------------------------------------------------------
// Types for gfx1250 WMMA (wave32): v_wmma_f32_16x16x32_bf16
// ---------------------------------------------------------------------------
typedef __attribute__((ext_vector_type(16))) __bf16          bf16x16;
typedef __attribute__((ext_vector_type(8)))  float           f32x8;
typedef __attribute__((ext_vector_type(8)))  unsigned short  ushort8;

union FragBF16 { bf16x16 v; ushort8 h[2]; };

// Exact operand types for the async-to-LDS builtin (from clang diagnostic):
//   arg0: v4i in global (AS1), arg1: v4i in LDS (AS3), arg2/3: imm offset/cpol.
typedef int v4i __attribute__((vector_size(4 * sizeof(int))));
typedef __attribute__((address_space(1))) v4i as1_v4i;
typedef __attribute__((address_space(3))) v4i as3_v4i;

#if defined(__has_builtin)
#if __has_builtin(__builtin_amdgcn_global_load_async_to_lds_b128)
#define HAS_ASYNC_LDS 1
#endif
#endif

#ifndef HAS_ASYNC_LDS
#define HAS_ASYNC_LDS 0
#endif

#if HAS_ASYNC_LDS
#if __has_builtin(__builtin_amdgcn_s_wait_asynccnt)
#define WAIT_ASYNC(n) __builtin_amdgcn_s_wait_asynccnt(n)
#else
#define WAIT_ASYNC(n) asm volatile("s_wait_asynccnt %0" ::"i"(n) : "memory")
#endif
#else
#define WAIT_ASYNC(n) ((void)0)
#endif

// Copy 16 bytes global -> LDS. Async (ASYNCcnt-tracked, no VGPR round trip)
// when the CDNA5 builtin is available; plain load/store otherwise.
__device__ __forceinline__ void copy16_g2l(const unsigned short* g,
                                           unsigned short* l) {
#if HAS_ASYNC_LDS
  __builtin_amdgcn_global_load_async_to_lds_b128((as1_v4i*)(void*)g,
                                                 (as3_v4i*)l, 0, 0);
#else
  *(ushort8*)l = *(const ushort8*)g;
#endif
}

__device__ __forceinline__ unsigned short f32_to_bf16(float f) {
  unsigned int u = __float_as_uint(f);
  unsigned int r = (u + 0x7FFFu + ((u >> 16) & 1u)) >> 16;  // RNE
  return (unsigned short)r;
}

// ---------------------------------------------------------------------------
// Pack KAN weights: Wb[OUT,IN] f32 + Ws[OUT,IN,8] f32 -> Wt[OUT][9*IN] bf16
//   k = c*IN + i : c==0 -> base (silu branch), c=1..8 -> spline coeff c-1
// Row-major (k contiguous) so GEMM B tiles are 16B row-chunk loads.
// ---------------------------------------------------------------------------
__global__ __launch_bounds__(256) void pack_w(
    const float* __restrict__ Wb, const float* __restrict__ Ws,
    unsigned short* __restrict__ Wt, int OUT, int IN) {
  long t = (long)blockIdx.x * 256 + threadIdx.x;
  if (t >= (long)OUT * IN) return;
  int i = (int)(t % IN);           // adjacent threads -> adjacent i: coalesced
  int o = (int)(t / IN);
  long base = (long)o * (9L * IN);
  Wt[base + i] = f32_to_bf16(Wb[(long)o * IN + i]);
#pragma unroll
  for (int c = 0; c < 8; ++c)
    Wt[base + (long)(c + 1) * IN + i] =
        f32_to_bf16(Ws[((long)o * IN + i) * 8 + c]);
}

// ---------------------------------------------------------------------------
// Spline expansion: X[N,IN] f32 -> Xe[N, 9*IN] bf16
//   Xe[n, 0*IN+i]     = silu(x)           (optionally relu'd input first)
//   Xe[n, (c+1)*IN+i] = B_c(x), c=0..7    (cubic Cox-de Boor, uniform grid)
// grid[t] = 0.4*t - 2.2 ; uniform knots => denominators are constants.
// ---------------------------------------------------------------------------
__global__ __launch_bounds__(256) void expand(
    const float* __restrict__ X, unsigned short* __restrict__ Xe,
    long total, int IN, int doRelu) {
  long t = (long)blockIdx.x * 256 + threadIdx.x;
  if (t >= total) return;
  long n = t / IN;
  int  i = (int)(t % IN);
  float x = X[t];
  if (doRelu) x = fmaxf(x, 0.0f);
  float sil = x / (1.0f + __expf(-x));
  long base = n * (long)(9 * IN);
  Xe[base + i] = f32_to_bf16(sil);

  float b[11];
#pragma unroll
  for (int t0 = 0; t0 < 11; ++t0) {
    float g0 = 0.4f * t0 - 2.2f;
    float g1 = 0.4f * (t0 + 1) - 2.2f;
    b[t0] = (x >= g0 && x < g1) ? 1.0f : 0.0f;
  }
#pragma unroll
  for (int j = 1; j <= 3; ++j) {
    float inv = 1.0f / (0.4f * j);
#pragma unroll
    for (int t0 = 0; t0 < 10; ++t0) {
      if (t0 < 11 - j) {
        float gl = 0.4f * t0 - 2.2f;
        float gr = 0.4f * (t0 + j + 1) - 2.2f;
        b[t0] = ((x - gl) * b[t0] + (gr - x) * b[t0 + 1]) * inv;
      }
    }
  }
#pragma unroll
  for (int c = 0; c < 8; ++c)
    Xe[base + (long)(c + 1) * IN + i] = f32_to_bf16(b[c]);
}

// ---------------------------------------------------------------------------
// Tiled WMMA GEMM: C[M,N] = A[M,K] * B^T (bf16 in, f32 accumulate)
//   A: [M][lda], k contiguous.   B: [N][ldb], k contiguous (weights row-major,
//   K-matrix rows, or transposed-V rows).  Both tile loads are 16B chunks ->
//   GLOBAL_LOAD_ASYNC_TO_LDS_B128 with double-buffered LDS pipeline.
// Block: 256 threads = 8 waves; tile 128(M) x 128(N); K step 32.
// Wave (wm 0..3, wn 0..1) owns 32x64 = 2x4 wmma fragments.
// Batched (z) with batch/head strides for attention reuse.
//   C_TRANSPOSE: store C[n][m] (used to produce V^T for the P*V GEMM).
// ---------------------------------------------------------------------------
template <bool OUT_BF16, bool C_TRANSPOSE>
__global__ __launch_bounds__(256) void kan_wmma_gemm(
    const unsigned short* __restrict__ Ag, long lda,
    const unsigned short* __restrict__ Bg, long ldb,
    void* __restrict__ Cg, long ldc, long K, float scale, int nh,
    long aBs, long aHs, long bBs, long bHs, long cBs, long cHs) {
  __shared__ unsigned short Ash[2][128][40];  // [buf][m][k] padded rows (16B)
  __shared__ unsigned short Bsh[2][128][40];  // [buf][n][k]

  const int tid = threadIdx.x;
  const int lane = tid & 31;
  const int wave = tid >> 5;
  const int wm = wave & 3;
  const int wn = wave >> 2;
  const int laneHalf = lane >> 4;
  const int laneMod = lane & 15;

  const long mT = (long)blockIdx.y * 128;
  const long nT = (long)blockIdx.x * 128;
  const int z = blockIdx.z;
  const int bb = z / nh, hh = z % nh;
  Ag += (long)bb * aBs + (long)hh * aHs;
  Bg += (long)bb * bBs + (long)hh * bHs;

  // Per-thread chunk coordinates: 512 16B chunks per 128x32 tile, 2 per thread.
  const int m0 = tid >> 2;              // 0..63   (chunk p=0)
  const int m1 = (tid + 256) >> 2;      // 64..127 (chunk p=1)
  const int kq0 = (tid & 3) * 8;        // 0,8,16,24

  auto issue_tile = [&](int buf, long k0) {
    copy16_g2l(Ag + (mT + m0) * lda + k0 + kq0, &Ash[buf][m0][kq0]);
    copy16_g2l(Ag + (mT + m1) * lda + k0 + kq0, &Ash[buf][m1][kq0]);
    copy16_g2l(Bg + (nT + m0) * ldb + k0 + kq0, &Bsh[buf][m0][kq0]);
    copy16_g2l(Bg + (nT + m1) * ldb + k0 + kq0, &Bsh[buf][m1][kq0]);
  };

  f32x8 acc[2][4] = {};

  int buf = 0;
  issue_tile(0, 0);  // prologue: 4 async ops in flight per wave

  for (long k0 = 0; k0 < K; k0 += 32) {
    const bool hasNext = (k0 + 32) < K;
    if (hasNext) issue_tile(buf ^ 1, k0 + 32);  // pipeline next tile
    if (hasNext) { WAIT_ASYNC(4); } else { WAIT_ASYNC(0); }  // tile t resident
    __syncthreads();

    // ---- fragments per documented gfx1250 layouts ----
    // A 16x32 bf16: lane l<16 row l, K {kH*8..} in v0-3, {kH*8+16..} in v4-7
    FragBF16 afr[2];
#pragma unroll
    for (int ms = 0; ms < 2; ++ms) {
      int row = wm * 32 + ms * 16 + laneMod;
      afr[ms].h[0] = *(const ushort8*)&Ash[buf][row][laneHalf * 8];
      afr[ms].h[1] = *(const ushort8*)&Ash[buf][row][laneHalf * 8 + 16];
    }
    // B 32x16 bf16: lane holds column laneMod, K {kH*8..} in v0-3, {kH*8+16..} in v4-7
#pragma unroll
    for (int ns = 0; ns < 4; ++ns) {
      int col = wn * 64 + ns * 16 + laneMod;
      FragBF16 bfr;
      bfr.h[0] = *(const ushort8*)&Bsh[buf][col][laneHalf * 8];
      bfr.h[1] = *(const ushort8*)&Bsh[buf][col][laneHalf * 8 + 16];
#pragma unroll
      for (int ms = 0; ms < 2; ++ms) {
        acc[ms][ns] = __builtin_amdgcn_wmma_f32_16x16x32_bf16(
            false, afr[ms].v, false, bfr.v, (short)0, acc[ms][ns], false, false);
      }
    }
    __syncthreads();  // all waves done reading buf before it is overwritten
    buf ^= 1;
  }

  // ---- epilogue: C/D layout VGPR j -> M = j + 8*laneHalf, N = laneMod ----
  long cOff = (long)bb * cBs + (long)hh * cHs;
#pragma unroll
  for (int ms = 0; ms < 2; ++ms) {
#pragma unroll
    for (int ns = 0; ns < 4; ++ns) {
      long row0 = mT + wm * 32 + ms * 16 + laneHalf * 8;
      long col = nT + wn * 64 + ns * 16 + laneMod;
#pragma unroll
      for (int j = 0; j < 8; ++j) {
        float v = acc[ms][ns][j] * scale;
        long idx = C_TRANSPOSE ? (cOff + col * ldc + (row0 + j))
                               : (cOff + (row0 + j) * ldc + col);
        if (OUT_BF16) ((unsigned short*)Cg)[idx] = f32_to_bf16(v);
        else          ((float*)Cg)[idx] = v;
      }
    }
  }
}

// ---------------------------------------------------------------------------
// Row softmax over S=1024 f32 scores, writing bf16 probs IN PLACE at the row
// base (u16 offset row*2S). Per-row aliasing only: all loads precede writes.
// ---------------------------------------------------------------------------
__global__ __launch_bounds__(256) void softmax_bf16_inplace(
    float* __restrict__ sc, int S) {
  long base = (long)blockIdx.x * S;
  int tid = threadIdx.x;
  __shared__ float red[256];
  float v[4];
#pragma unroll
  for (int i = 0; i < 4; ++i) v[i] = sc[base + tid + i * 256];
  float m = fmaxf(fmaxf(v[0], v[1]), fmaxf(v[2], v[3]));
  red[tid] = m; __syncthreads();
  for (int o = 128; o > 0; o >>= 1) {
    if (tid < o) red[tid] = fmaxf(red[tid], red[tid + o]);
    __syncthreads();
  }
  float rowMax = red[0]; __syncthreads();
  float e[4], s = 0.0f;
#pragma unroll
  for (int i = 0; i < 4; ++i) { e[i] = __expf(v[i] - rowMax); s += e[i]; }
  red[tid] = s; __syncthreads();
  for (int o = 128; o > 0; o >>= 1) {
    if (tid < o) red[tid] += red[tid + o];
    __syncthreads();
  }
  float inv = 1.0f / red[0];
  __syncthreads();
  unsigned short* out = (unsigned short*)(sc + base);
#pragma unroll
  for (int i = 0; i < 4; ++i) out[tid + i * 256] = f32_to_bf16(e[i] * inv);
}

// ---------------------------------------------------------------------------
// Residual + LayerNorm (in place on y = o-proj output): y = LN(y + x)*w + b
// ---------------------------------------------------------------------------
__global__ __launch_bounds__(256) void resid_ln(
    float* __restrict__ y, const float* __restrict__ x,
    const float* __restrict__ w, const float* __restrict__ b) {
  int tid = threadIdx.x;
  long base = (long)blockIdx.x * 512;
  __shared__ float red[256];
  float v0 = y[base + tid] + x[base + tid];
  float v1 = y[base + tid + 256] + x[base + tid + 256];
  red[tid] = v0 + v1; __syncthreads();
  for (int o = 128; o > 0; o >>= 1) {
    if (tid < o) red[tid] += red[tid + o];
    __syncthreads();
  }
  float mean = red[0] * (1.0f / 512.0f); __syncthreads();
  float d0 = v0 - mean, d1 = v1 - mean;
  red[tid] = d0 * d0 + d1 * d1; __syncthreads();
  for (int o = 128; o > 0; o >>= 1) {
    if (tid < o) red[tid] += red[tid + o];
    __syncthreads();
  }
  float inv = rsqrtf(red[0] * (1.0f / 512.0f) + 1e-5f);
  y[base + tid]       = d0 * inv * w[tid] + b[tid];
  y[base + tid + 256] = d1 * inv * w[tid + 256] + b[tid + 256];
}

// ---------------------------------------------------------------------------
extern "C" void kernel_launch(void* const* d_in, const int* in_sizes, int n_in,
                              void* d_out, int out_size, void* d_ws, size_t ws_size,
                              hipStream_t stream) {
  (void)in_sizes; (void)n_in; (void)out_size; (void)ws_size;
  const float* x   = (const float*)d_in[0];
  const float* k1b = (const float*)d_in[1];  const float* k1s = (const float*)d_in[2];
  const float* k2b = (const float*)d_in[3];  const float* k2s = (const float*)d_in[4];
  const float* qbw = (const float*)d_in[5];  const float* qsw = (const float*)d_in[6];
  const float* kbw = (const float*)d_in[7];  const float* ksw = (const float*)d_in[8];
  const float* vbw = (const float*)d_in[9];  const float* vsw = (const float*)d_in[10];
  const float* obw = (const float*)d_in[11]; const float* osw = (const float*)d_in[12];
  const float* lnw = (const float*)d_in[13]; const float* lnb = (const float*)d_in[14];

  const long N = 8192, D = 512, HD = 1024, S = 1024, NH = 4;

  // -------- workspace carving (256B aligned) --------
  char* w = (char*)d_ws;
  size_t off = 0;
  auto carve = [&](size_t bytes) -> char* {
    char* p = w + off; off += (bytes + 255) & ~(size_t)255; return p;
  };
  unsigned short* wt1 = (unsigned short*)carve((size_t)9 * 512 * 1024 * 2);
  unsigned short* wt2 = (unsigned short*)carve((size_t)9 * 1024 * 512 * 2);
  unsigned short* wtq = (unsigned short*)carve((size_t)9 * 512 * 512 * 2);
  unsigned short* wtk = (unsigned short*)carve((size_t)9 * 512 * 512 * 2);
  unsigned short* wtv = (unsigned short*)carve((size_t)9 * 512 * 512 * 2);
  unsigned short* wto = (unsigned short*)carve((size_t)9 * 512 * 512 * 2);
  // Xe (151MB) and attention scores (134MB) have disjoint live ranges: overlay.
  char* big = carve((size_t)N * 9216 * 2);
  unsigned short* xe = (unsigned short*)big;
  float* scoresF = (float*)big;                 // [32][1024][1024] f32
  unsigned short* probs = (unsigned short*)big; // bf16 in-place, row stride 2048 u16
  float* h1  = (float*)carve((size_t)N * HD * 4);
  float* h2  = (float*)carve((size_t)N * D * 4);
  unsigned short* qb  = (unsigned short*)carve((size_t)N * D * 2);
  unsigned short* kb  = (unsigned short*)carve((size_t)N * D * 2);
  unsigned short* vbT = (unsigned short*)carve((size_t)D * N * 2);  // V^T [512][8192]
  float* att = (float*)carve((size_t)N * D * 4);
  float* yo  = (float*)d_out;                   // o-proj output written in place

  // -------- pack all KAN weights to bf16, [OUT][9*IN] row-major --------
  pack_w<<<(1024 * 512 + 255) / 256, 256, 0, stream>>>(k1b, k1s, wt1, 1024, 512);
  pack_w<<<(512 * 1024 + 255) / 256, 256, 0, stream>>>(k2b, k2s, wt2, 512, 1024);
  pack_w<<<(512 * 512 + 255) / 256, 256, 0, stream>>>(qbw, qsw, wtq, 512, 512);
  pack_w<<<(512 * 512 + 255) / 256, 256, 0, stream>>>(kbw, ksw, wtk, 512, 512);
  pack_w<<<(512 * 512 + 255) / 256, 256, 0, stream>>>(vbw, vsw, wtv, 512, 512);
  pack_w<<<(512 * 512 + 255) / 256, 256, 0, stream>>>(obw, osw, wto, 512, 512);

  // -------- KAN1: h1 = xe(x) @ wt1^T   [8192,1024], K=4608 --------
  expand<<<(int)(N * 512 / 256), 256, 0, stream>>>(x, xe, N * 512, 512, 0);
  kan_wmma_gemm<false, false><<<dim3(8, 64, 1), 256, 0, stream>>>(
      xe, 4608, wt1, 4608, h1, 1024, 4608, 1.0f, 1, 0, 0, 0, 0, 0, 0);

  // -------- KAN2: h2 = xe(relu(h1)) @ wt2^T   [8192,512], K=9216 --------
  expand<<<(int)(N * 1024 / 256), 256, 0, stream>>>(h1, xe, N * 1024, 1024, 1);
  kan_wmma_gemm<false, false><<<dim3(4, 64, 1), 256, 0, stream>>>(
      xe, 9216, wt2, 9216, h2, 512, 9216, 1.0f, 1, 0, 0, 0, 0, 0, 0);

  // -------- q,k,v KAN projections (bf16 outputs), K=4608 --------
  expand<<<(int)(N * 512 / 256), 256, 0, stream>>>(h2, xe, N * 512, 512, 0);
  kan_wmma_gemm<true, false><<<dim3(4, 64, 1), 256, 0, stream>>>(
      xe, 4608, wtq, 4608, qb, 512, 4608, 1.0f, 1, 0, 0, 0, 0, 0, 0);
  kan_wmma_gemm<true, false><<<dim3(4, 64, 1), 256, 0, stream>>>(
      xe, 4608, wtk, 4608, kb, 512, 4608, 1.0f, 1, 0, 0, 0, 0, 0, 0);
  // V stored transposed so the P*V GEMM's B tiles are contiguous-k rows.
  kan_wmma_gemm<true, true><<<dim3(4, 64, 1), 256, 0, stream>>>(
      xe, 4608, wtv, 4608, vbT, 8192, 4608, 1.0f, 1, 0, 0, 0, 0, 0, 0);

  // -------- attention scores: Q K^T / sqrt(128) per (b,h), K=dh=128 --------
  kan_wmma_gemm<false, false><<<dim3(8, 8, 32), 256, 0, stream>>>(
      qb, 512, kb, 512, scoresF, 1024, 128, 0.08838834764831845f, (int)NH,
      S * 512, 128,           // A (Q) batch/head strides, u16 elems
      S * 512, 128,           // B (K)
      4 * S * S, S * S);      // C (scores) f32 elems

  // -------- softmax (f32 -> bf16 probs in place) --------
  softmax_bf16_inplace<<<32 * 1024, 256, 0, stream>>>(scoresF, (int)S);

  // -------- P @ V per (b,h): [1024,128], K=1024 --------
  kan_wmma_gemm<false, false><<<dim3(1, 8, 32), 256, 0, stream>>>(
      probs, 2048, vbT, 8192, att, 512, 1024, 1.0f, (int)NH,
      4 * S * 2048, S * 2048,  // A (probs) strides, u16 elems
      1024, 128 * 8192,        // B (V^T): per-batch token offset / per-head rows
      S * 512, 128);           // C (att out) f32

  // -------- output KAN projection -> d_out, K=4608 --------
  expand<<<(int)(N * 512 / 256), 256, 0, stream>>>(att, xe, N * 512, 512, 0);
  kan_wmma_gemm<false, false><<<dim3(4, 64, 1), 256, 0, stream>>>(
      xe, 4608, wto, 4608, yo, 512, 4608, 1.0f, 1, 0, 0, 0, 0, 0, 0);

  // -------- residual + LayerNorm in place on d_out --------
  resid_ln<<<(int)N, 256, 0, stream>>>(yo, x, lnw, lnb);
}
